// Geo_Aware_Attn_23192823398862
// MI455X (gfx1250) — compile-verified, matrix operand order checked by reading, not run on hardware
//
#include <hip/hip_runtime.h>
#include <hip/hip_bf16.h>
#include <math.h>

// ---------------- CDNA5 WMMA / TDM types ----------------
typedef __attribute__((ext_vector_type(16))) _Float16     v16h;
typedef __attribute__((ext_vector_type(8)))  float        v8f;
typedef __attribute__((ext_vector_type(4)))  unsigned int v4u;
typedef __attribute__((ext_vector_type(8)))  int          v8i;
typedef __attribute__((ext_vector_type(4)))  int          v4i;

// ---------------- problem constants ----------------
#define BDIM 4
#define DDIM 32
#define BD   128           // B*D
#define NN   768           // tokens
#define LL   192           // 2*seq_len
#define DM   128           // d_model
#define PL   24            // pred_len
#define IND  3             // in_dim
#define NM   32            // N_MODEL

#define LNEG        (-23.0258509299404568f)   // -ln(1e10)
#define SCALE_TEMP  (1.7677669529663689f)     // (1/sqrt(128)) / 0.05
#define LNEG_TEMP   (LNEG * 20.0f)

// LDS strides (elements)
#define XLD 200
#define WLD 200
#define OLD 136            // staging stride for coalesced q/k store (row = 272B, 16B-aligned)
#define QLD 132            // 128 + 4 halves pad (TDM pad: 2 DWORDs per 64-DWORD row)
#define SLD 772

#define QKV_SMEM  (64*XLD*2 + 128*WLD*2)                       // 76,800 B
#define ATTN_SMEM (64*SLD*4 + 64*QLD*2)                        // 214,528 B
#define QS_LDS_OFF (64*SLD*4)                                  // byte offset of qs in dynamic LDS

#if __has_builtin(__builtin_amdgcn_tensor_load_to_lds) && __has_builtin(__builtin_amdgcn_s_wait_tensorcnt)
#define HAVE_TDM 1
#endif

// ---------------- fragment helpers (ISA 7.12.2 layouts, wave32) ----------------
// A(16x32 f16) / B(32x16 f16 stored as Mat[n][k], i.e. transposed row-major):
// lane l: row = l&15, kbase = (l>>4)*8 ; vgpr v: k = (v>>2)*16 + kbase + (v&3)*2
// => per lane: two contiguous 8-half (16B) chunks at k = kbase and k = 16+kbase.
__device__ __forceinline__ v16h frag_ld_h(const _Float16* s, int ld) {
  const int lane = threadIdx.x & 31;
  const int r = lane & 15;
  const int kg = (lane >> 4) << 3;
  v16h f;
#pragma unroll
  for (int v = 0; v < 8; ++v) {
    const int k = ((v >> 2) << 4) + kg + ((v & 3) << 1);
    f[2*v]   = s[r*ld + k];
    f[2*v+1] = s[r*ld + k + 1];
  }
  return f;
}

// A-frag sourced from f32 LDS (converted to f16 on the fly)
__device__ __forceinline__ v16h frag_ld_f32(const float* s, int ld) {
  const int lane = threadIdx.x & 31;
  const int r = lane & 15;
  const int kg = (lane >> 4) << 3;
  v16h f;
#pragma unroll
  for (int v = 0; v < 8; ++v) {
    const int k = ((v >> 2) << 4) + kg + ((v & 3) << 1);
    f[2*v]   = (_Float16)s[r*ld + k];
    f[2*v+1] = (_Float16)s[r*ld + k + 1];
  }
  return f;
}

__device__ __forceinline__ v8f wmma_f16(v16h a, v16h b, v8f c) {
  return __builtin_amdgcn_wmma_f32_16x16x32_f16(false, a, false, b, (short)0, c, false, false);
}

// ---------------- geo pipeline ----------------
__global__ __launch_bounds__(256) void k_geo0(const float* __restrict__ cpan,
                                              const float* __restrict__ csta,
                                              const float* __restrict__ W_sta,
                                              const float* __restrict__ b_sta,
                                              float* __restrict__ cn,
                                              float* __restrict__ nodevec) {
  __shared__ float r0[256], r1[256], r2[256], r3[256];
  const int tid = threadIdx.x;
  float mnLa = 3.4e38f, mxLa = -3.4e38f, mnLo = 3.4e38f, mxLo = -3.4e38f;
  for (int i = tid; i < BDIM*NN; i += 256) {
    float la = cpan[2*i], lo = cpan[2*i+1];
    mnLa = fminf(mnLa, la); mxLa = fmaxf(mxLa, la);
    mnLo = fminf(mnLo, lo); mxLo = fmaxf(mxLo, lo);
  }
  r0[tid] = mnLa; r1[tid] = mxLa; r2[tid] = mnLo; r3[tid] = mxLo;
  __syncthreads();
  for (int off = 128; off > 0; off >>= 1) {
    if (tid < off) {
      r0[tid] = fminf(r0[tid], r0[tid+off]);
      r1[tid] = fmaxf(r1[tid], r1[tid+off]);
      r2[tid] = fminf(r2[tid], r2[tid+off]);
      r3[tid] = fmaxf(r3[tid], r3[tid+off]);
    }
    __syncthreads();
  }
  const float minLa = r0[0], minLo = r2[0];
  const float rgLa = r1[0] - r0[0] + 1e-8f;
  const float rgLo = r3[0] - r2[0] + 1e-8f;
  for (int i = tid; i < NN; i += 256) {
    cn[2*i]   = (csta[2*i]   - minLa) / rgLa;
    cn[2*i+1] = (csta[2*i+1] - minLo) / rgLo;
  }
  __syncthreads();
  for (int i = tid; i < NN*NM; i += 256) {
    int n = i >> 5, j = i & 31;
    nodevec[i] = cn[2*n]*W_sta[2*j] + cn[2*n+1]*W_sta[2*j+1] + b_sta[j];
  }
}

__global__ __launch_bounds__(256) void k_geo1(const float* __restrict__ nodevec,
                                              const float* __restrict__ W_n1,
                                              const float* __restrict__ b_n1,
                                              const float* __restrict__ W_n2,
                                              const float* __restrict__ b_n2,
                                              float* __restrict__ nv1,
                                              float* __restrict__ nv2) {
  __shared__ float sq[256];
  const int tid = threadIdx.x;
  const int n = blockIdx.x*8 + (tid >> 5);
  const int j = tid & 31;
  float s1 = b_n1[j], s2 = b_n2[j];
  for (int t = 0; t < NM; ++t) {
    float nv = nodevec[n*NM + t];
    float sn = sinf(nv), cs = cosf(nv);
    s1 += sn*W_n1[j*96 + t] + cs*W_n1[j*96 + 32 + t] + nv*W_n1[j*96 + 64 + t];
    s2 += sn*W_n2[j*96 + t] + cs*W_n2[j*96 + 32 + t] + nv*W_n2[j*96 + 64 + t];
  }
  float a1 = fmaxf(s1, 0.0f), a2 = fmaxf(s2, 0.0f);
  sq[tid] = a1*a1;
  __syncthreads();
  for (int off = 16; off > 0; off >>= 1) { if ((tid & 31) < off) sq[tid] += sq[tid+off]; __syncthreads(); }
  float norm1 = sqrtf(sq[tid & ~31]);
  __syncthreads();
  sq[tid] = a2*a2;
  __syncthreads();
  for (int off = 16; off > 0; off >>= 1) { if ((tid & 31) < off) sq[tid] += sq[tid+off]; __syncthreads(); }
  float norm2 = sqrtf(sq[tid & ~31]);
  nv1[n*NM + j] = a1 / fmaxf(norm1, 1e-12f);
  nv2[n*NM + j] = a2 / fmaxf(norm2, 1e-12f);
}

__global__ __launch_bounds__(256) void k_geo2(const float* __restrict__ cn,
                                              const float* __restrict__ nv1,
                                              const float* __restrict__ nv2,
                                              const float* __restrict__ dsc,
                                              float* __restrict__ geo) {
  const int idx = blockIdx.x*256 + threadIdx.x;
  const int i = idx / NN, j = idx % NN;
  float d0 = cn[2*i] - cn[2*j], d1 = cn[2*i+1] - cn[2*j+1];
  float dist = sqrtf(d0*d0 + d1*d1 + 1e-8f);
  float dscale = 3.0f / (1.0f + __expf(-dsc[0]));
  float dsim = __expf(-dist * dscale);
  float g1 = 0.0f, g2 = 0.0f;
#pragma unroll 8
  for (int t = 0; t < NM; ++t) {
    g1 += nv1[i*NM + t] * nv2[j*NM + t];
    g2 += nv1[j*NM + t] * nv2[i*NM + t];
  }
  geo[idx] = 0.3f*(g1 + g2) + 0.4f*dsim + 0.05f;
}

// ---------------- QKV projection (WMMA f16); V stored transposed ----------------
__global__ __launch_bounds__(256) void k_qkv(const float* __restrict__ x,
                                             const float* __restrict__ Wq, const float* __restrict__ bq,
                                             const float* __restrict__ Wk, const float* __restrict__ bk,
                                             const float* __restrict__ Wv, const float* __restrict__ bv,
                                             _Float16* __restrict__ qh, _Float16* __restrict__ kh,
                                             _Float16* __restrict__ vt) {
  extern __shared__ char smem[];
  _Float16* xs  = (_Float16*)smem;            // 64 x XLD
  _Float16* wsh = xs + 64*XLD;                // 128 x WLD (also reused as 64 x OLD out-tile)
  const int bd = blockIdx.y;
  const int r0 = blockIdx.x * 64;
  const int tid = threadIdx.x;
  const int wave = tid >> 5;
  const int lane = tid & 31;
  const int n = lane & 15, mh = (lane >> 4) * 8;

  for (int i = tid; i < 64*LL; i += 256) {
    int r = i / LL, c = i % LL;
    xs[r*XLD + c] = (_Float16)x[((size_t)bd*NN + r0 + r)*LL + c];
  }
  const float* Ws[3] = {Wq, Wk, Wv};
  const float* Bs[3] = {bq, bk, bv};

  for (int m = 0; m < 3; ++m) {
    __syncthreads();                          // out-tile copy of previous m done
    for (int i = tid; i < DM*LL; i += 256) {
      int r = i / LL, c = i % LL;
      wsh[r*WLD + c] = (_Float16)Ws[m][i];
    }
    __syncthreads();
    v8f accs[4];
#pragma unroll
    for (int s = 0; s < 4; ++s) {
      int t  = wave + (s << 3);
      int rt = t & 3, ct = t >> 2;            // 4 row-tiles x 8 col-tiles
      v8f acc = {};
#pragma unroll
      for (int kk = 0; kk < LL; kk += 32) {
        v16h a = frag_ld_h(xs  + (rt*16)*XLD + kk, XLD);
        v16h b = frag_ld_h(wsh + (ct*16)*WLD + kk, WLD);   // B(k,n)=W[n][k]
        acc = wmma_f16(a, b, acc);
      }
      accs[s] = acc;
    }
    if (m < 2) {
      // bounce through LDS so the global store is coalesced b128
      __syncthreads();                        // all B-frag reads of wsh complete
      _Float16* ot = wsh;                     // 64 x OLD f16
#pragma unroll
      for (int s = 0; s < 4; ++s) {
        int t = wave + (s << 3);
        int rt = t & 3, ct = t >> 2;
        float bias = Bs[m][ct*16 + n];
#pragma unroll
        for (int v = 0; v < 8; ++v)
          ot[(rt*16 + mh + v)*OLD + ct*16 + n] = (_Float16)(accs[s][v] + bias);
      }
      __syncthreads();
      _Float16* dst = (m == 0 ? qh : kh) + ((size_t)bd*NN + r0)*DM;
      for (int c8 = tid; c8 < (64*DM)/8; c8 += 256) {
        int off = c8 * 8, row = off >> 7, col = off & 127;
        _Float16* dp = dst + row*DM + col;
        const _Float16* sp = ot + row*OLD + col;
#pragma unroll
        for (int u = 0; u < 8; ++u) dp[u] = sp[u];
      }
    } else {
      // V transposed: vt[bd][dm][token] -> 8 consecutive f16 per lane (b128 store)
#pragma unroll
      for (int s = 0; s < 4; ++s) {
        int t = wave + (s << 3);
        int rt = t & 3, ct = t >> 2;
        float bias = Bs[m][ct*16 + n];
        _Float16* dst = vt + ((size_t)bd*DM + ct*16 + n)*NN + r0 + rt*16 + mh;
#pragma unroll
        for (int v = 0; v < 8; ++v) dst[v] = (_Float16)(accs[s][v] + bias);
      }
    }
  }
}

// ---------------- fused attention strip: scores -> softmax -> A -> A@v -> fc1 ----------------
__global__ __launch_bounds__(256) void k_attn(const _Float16* __restrict__ qh,
                                              const _Float16* __restrict__ kh,
                                              const _Float16* __restrict__ vt,
                                              const float* __restrict__ geo,
                                              const float* __restrict__ Wfc1,
                                              const float* __restrict__ bfc1,
                                              float* __restrict__ Aout,
                                              float* __restrict__ out1) {
  extern __shared__ char smem[];
  float*    st = (float*)smem;                        // 64 x SLD  scores/probs/t
  _Float16* qs = (_Float16*)(smem + QS_LDS_OFF);      // 64 x QLD
  const int bd = blockIdx.y;
  const int r0 = blockIdx.x * 64;
  const int tid = threadIdx.x;
  const int wave = tid >> 5;
  const int lane = tid & 31;

  // prefetch the geo strip this block will consume (global_prefetch_b8)
  {
    const float* base = geo + (size_t)r0*NN;
    for (int pfo = tid*32; pfo < 64*NN; pfo += 256*32)
      __builtin_prefetch(base + pfo, 0, 1);
  }

  // ---- stage Q strip (f16) ----
#ifdef HAVE_TDM
  // Tensor Data Mover: 2D tile 128x64 f16, row stride 128, LDS pad 2 DWORDs per
  // 64-DWORD row => effective LDS row stride = 132 halves (QLD).
  if (wave == 0) {
    unsigned long long ga = (unsigned long long)(const void*)(qh + ((size_t)bd*NN + r0)*DM);
    unsigned la = (unsigned)(size_t)(void*)qs;
    v4u g0;
    g0[0] = 1u;                                              // count=1, user mode
    g0[1] = la;                                              // lds_addr (bytes)
    g0[2] = (unsigned)(ga & 0xFFFFFFFFu);                    // global_addr[31:0]
    g0[3] = (unsigned)((ga >> 32) & 0x01FFFFFFu) | (2u << 30); // addr[56:32] | type=2
    v8i g1;
    g1[0] = (int)((1u<<16) | (1u<<20) | (5u<<22) | (1u<<25)); // data_size=2B, pad_en, intv=64DW, amt=2DW
    g1[1] = (int)(128u << 16);   // tensor_dim0[15:0]=128
    g1[2] = (int)(64u  << 16);   // tensor_dim0 hi=0 | tensor_dim1[15:0]=64
    g1[3] = (int)(128u << 16);   // tensor_dim1 hi=0 | tile_dim0=128
    g1[4] = (int)(64u);          // tile_dim1=64 | tile_dim2=0
    g1[5] = (int)(128u);         // tensor_dim0_stride lo = 128
    g1[6] = 0;                   // stride hi | tensor_dim1_stride lo
    g1[7] = 0;
    v4i gz4 = {0, 0, 0, 0};
    v8i gz8 = {0, 0, 0, 0, 0, 0, 0, 0};
    __builtin_amdgcn_tensor_load_to_lds(g0, g1, gz4, gz4, gz8, 0);
    __builtin_amdgcn_s_wait_tensorcnt(0);
  }
  __syncthreads();
#else
  for (int i = tid; i < 64*DM; i += 256) {
    int r = i >> 7, c = i & 127;
    qs[r*QLD + c] = qh[((size_t)bd*NN + r0 + r)*DM + c];
  }
  __syncthreads();
#endif

  // ---- scores: 64x768 strip = 4x48 tiles, 24 tiles per wave, K=128 ----
  for (int s = 0; s < 24; ++s) {
    int t  = wave + (s << 3);
    int rt = t & 3, ct = t >> 2;
    v8f acc = {};
#pragma unroll
    for (int kk = 0; kk < DM; kk += 32) {
      v16h a = frag_ld_h(qs + (rt*16)*QLD + kk, QLD);
      v16h b = frag_ld_h(kh + ((size_t)bd*NN + ct*16)*DM + kk, DM);  // B(k,n)=K[n][k]
      acc = wmma_f16(a, b, acc);
    }
    int n = lane & 15, mh = (lane >> 4) * 8;
    float* dst = st + (rt*16)*SLD + ct*16;
#pragma unroll
    for (int v = 0; v < 8; ++v) dst[(mh+v)*SLD + n] = acc[v];
  }
  __syncthreads();

  // ---- softmax: one wave per 8 rows; 32 lanes stride each row (coalesced) ----
  {
    const int rbase = wave * 8;
    for (int rr = 0; rr < 8; ++rr) {
      int row = rbase + rr;
      int gr = r0 + row;
      float* srow = st + row*SLD;
      const float* grow = geo + (size_t)gr*NN;
      float vals[24];
      float mx = -3.4e38f;
#pragma unroll
      for (int j = 0; j < 24; ++j) {
        int c = lane + 32*j;
        float g = grow[c];
        float v = srow[c]*SCALE_TEMP*g + (g == 0.0f ? LNEG_TEMP : 0.0f);
        vals[j] = v;
        mx = fmaxf(mx, v);
      }
#pragma unroll
      for (int off = 16; off > 0; off >>= 1) mx = fmaxf(mx, __shfl_xor(mx, off, 32));
      float sum = 0.0f;
#pragma unroll
      for (int j = 0; j < 24; ++j) {
        float e = __expf(vals[j] - mx);
        vals[j] = e;
        sum += e;
      }
#pragma unroll
      for (int off = 16; off > 0; off >>= 1) sum += __shfl_xor(sum, off, 32);
      float rinv = 1.0f / sum;
      float* arow = Aout + ((size_t)bd*NN + gr)*NN;
#pragma unroll
      for (int j = 0; j < 24; ++j) {
        int c = lane + 32*j;
        float p = vals[j] * rinv;
        srow[c] = p;
        __builtin_nontemporal_store(p, arow + c);   // A: streamed, written once
      }
    }
  }
  __syncthreads();

  // ---- t = A @ v : 64x128 = 4x8 tiles, 4 per wave, K=768 ----
  // B-frags from transposed V (vt[dm][token]) -> contiguous 16B chunks per lane.
  v8f tacc[4];
#pragma unroll
  for (int s = 0; s < 4; ++s) {
    int t  = wave + (s << 3);
    int rt = t & 3, ct = t >> 2;
    v8f acc = {};
    for (int kk = 0; kk < NN; kk += 32) {
      v16h a = frag_ld_f32(st + (rt*16)*SLD + kk, SLD);
      v16h b = frag_ld_h(vt + ((size_t)bd*DM + ct*16)*NN + kk, NN);  // B(k,n)=vt[n][k]
      acc = wmma_f16(a, b, acc);
    }
    tacc[s] = acc;
  }
  __syncthreads();                        // probs no longer needed
#pragma unroll
  for (int s = 0; s < 4; ++s) {
    int t  = wave + (s << 3);
    int rt = t & 3, ct = t >> 2;
    int n = lane & 15, mh = (lane >> 4) * 8;
    float* dst = st + (rt*16)*DM + ct*16; // reuse st as 64x128 f32
#pragma unroll
    for (int v = 0; v < 8; ++v) dst[(mh+v)*DM + n] = tacc[s][v];
  }
  __syncthreads();

  // ---- fc1: (64x128) @ (128x24)^T + bias -> out1 ----
  for (int i = tid; i < 64*PL; i += 256) {
    int row = i / PL, p = i % PL;
    float sum = bfc1[p];
    const float* tr = st + row*DM;
    const float* wr = Wfc1 + p*DM;
#pragma unroll 8
    for (int c = 0; c < DM; ++c) sum += tr[c] * wr[c];
    out1[((size_t)bd*NN + r0 + row)*PL + p] = sum;
  }
}

// ---------------- final fc2 + transposes ----------------
__global__ __launch_bounds__(256) void k_out2(const float* __restrict__ out1,
                                              const float* __restrict__ Wfc2,
                                              const float* __restrict__ bfc2,
                                              float* __restrict__ out) {
  const int idx = blockIdx.x*256 + threadIdx.x;
  if (idx >= BDIM*IND*NN*PL) return;
  int p = idx % PL;
  int n = (idx / PL) % NN;
  int i = (idx / (PL*NN)) % IND;
  int b = idx / (PL*NN*IND);
  float sum = bfc2[i];
#pragma unroll 8
  for (int d = 0; d < DDIM; ++d)
    sum += out1[(((size_t)(b*DDIM + d))*NN + n)*PL + p] * Wfc2[i*DDIM + d];
  out[idx] = sum;   // layout (B, in_dim, N, pred_len)
}

// ---------------- launch ----------------
extern "C" void kernel_launch(void* const* d_in, const int* in_sizes, int n_in,
                              void* d_out, int out_size, void* d_ws, size_t ws_size,
                              hipStream_t stream) {
  (void)in_sizes; (void)n_in; (void)out_size; (void)ws_size;
  const float* x     = (const float*)d_in[1];
  const float* csta  = (const float*)d_in[2];
  const float* cpan  = (const float*)d_in[3];
  const float* Wq    = (const float*)d_in[4];  const float* bq    = (const float*)d_in[5];
  const float* Wk    = (const float*)d_in[6];  const float* bk    = (const float*)d_in[7];
  const float* Wv    = (const float*)d_in[8];  const float* bv    = (const float*)d_in[9];
  const float* W_sta = (const float*)d_in[10]; const float* b_sta = (const float*)d_in[11];
  const float* W_n1  = (const float*)d_in[12]; const float* b_n1  = (const float*)d_in[13];
  const float* W_n2  = (const float*)d_in[14]; const float* b_n2  = (const float*)d_in[15];
  const float* dsc   = (const float*)d_in[16];
  const float* Wfc1  = (const float*)d_in[17]; const float* bfc1  = (const float*)d_in[18];
  const float* Wfc2  = (const float*)d_in[19]; const float* bfc2  = (const float*)d_in[20];

  // workspace layout
  _Float16* qh = (_Float16*)d_ws;
  _Float16* kh = qh + (size_t)BD*NN*DM;
  _Float16* vt = kh + (size_t)BD*NN*DM;       // transposed V: [bd][dm][token]
  float* cn      = (float*)(vt + (size_t)BD*NN*DM);
  float* nodevec = cn + NN*2;
  float* nv1     = nodevec + NN*NM;
  float* nv2     = nv1 + NN*NM;
  float* out1    = nv2 + NN*NM;

  // output layout: out (221184) | A (128*768*768) | geo (768*768)
  float* outp = (float*)d_out;
  float* Aout = outp + (size_t)BDIM*IND*NN*PL;
  float* geo  = Aout + (size_t)BD*NN*NN;

  k_geo0<<<1, 256, 0, stream>>>(cpan, csta, W_sta, b_sta, cn, nodevec);
  k_geo1<<<NN/8, 256, 0, stream>>>(nodevec, W_n1, b_n1, W_n2, b_n2, nv1, nv2);
  k_geo2<<<(NN*NN)/256, 256, 0, stream>>>(cn, nv1, nv2, dsc, geo);
  k_qkv<<<dim3(NN/64, BD), 256, QKV_SMEM, stream>>>(x, Wq, bq, Wk, bk, Wv, bv, qh, kh, vt);
  k_attn<<<dim3(NN/64, BD), 256, ATTN_SMEM, stream>>>(qh, kh, vt, geo, Wfc1, bfc1, Aout, out1);
  k_out2<<<(BDIM*IND*NN*PL + 255)/256, 256, 0, stream>>>(out1, Wfc2, bfc2, outp);
}
